// ReducedGammaSpectralFilter_30227979829453
// MI455X (gfx1250) — compile-verified, matrix operand order checked by reading, not run on hardware
//
#include <hip/hip_runtime.h>
#include <hip/hip_bf16.h>
#include <stdint.h>

// Problem constants (match reference)
#define BSZ 64
#define DD  1024
#define PP  512
#define KK  64
#define NC  576      // P + K (concatenated X | Xq)
#define NCP 640      // NC padded to a multiple of 128 (rows 576..639 zero)
#define LSTEPS 24

// GEMM tiling: 128x128 block tile, 8 waves, each wave 32(m) x 64(n)
#define BM 128
#define BN 128
#define BK 32
#define LDS_STRIDE 40   // 32 + 8 halfword pad -> conflict-free frag reads

typedef __attribute__((ext_vector_type(16))) __bf16 v16bf;
typedef __attribute__((ext_vector_type(8)))  float  v8f;
typedef __attribute__((ext_vector_type(8)))  int    v8i;

union FragAB { v8i i; v16bf b; };
union Pack8  { int4 v; unsigned short u[8]; };
union Pack4  { uint2 v; unsigned short u[4]; };

__device__ __forceinline__ unsigned short f32_to_bf16(float f) {
    uint32_t u = __float_as_uint(f);
    uint32_t r = u + 0x7FFFu + ((u >> 16) & 1u);   // round to nearest even
    return (unsigned short)(r >> 16);
}
__device__ __forceinline__ v8f v8f_zero() {
    v8f z;
#pragma unroll
    for (int i = 0; i < 8; ++i) z[i] = 0.0f;
    return z;
}

__device__ __forceinline__ v8f wmma_bf16(v16bf a, v16bf b, v8f c) {
    return __builtin_amdgcn_wmma_f32_16x16x32_bf16(false, a, false, b, (short)0, c,
                                                   false, false);
}

// Wave-relative LDS byte offset of a __shared__ object (generic -> AS(3) -> u32)
__device__ __forceinline__ unsigned lds_off_of(const void* p) {
    return (unsigned)(unsigned long long)(const __attribute__((address_space(3))) void*)p;
}

// ---- CDNA5 async tile staging: global -> LDS DMA, tracked by ASYNCcnt ----
// 128 rows x 32 bf16 per tile = 512 chunks of 16B; 256 threads x 2 chunks.
__device__ __forceinline__ void stage_async(const unsigned short* __restrict__ g,
                                            int row0, int kk,
                                            unsigned short* lds, int tid) {
#pragma unroll
    for (int i = 0; i < 2; ++i) {
        int c   = tid + i * 256;
        int row = c >> 2;
        int k8  = (c & 3) * 8;
        unsigned long long ga =
            (unsigned long long)(g + (size_t)(row0 + row) * DD + kk + k8);
        unsigned la = lds_off_of(lds + row * LDS_STRIDE + k8);
        asm volatile("global_load_async_to_lds_b128 %0, %1, off"
                     :: "v"(la), "v"(ga) : "memory");
    }
}
__device__ __forceinline__ void wait_async_all() {
    asm volatile("s_wait_asynccnt 0x0" ::: "memory");
}

// Load a 16x32 bf16 fragment from an LDS tile (rows contiguous in k).
// ISA A-layout: lane l -> row = l&15, K in [8*(l>>4), +8) (v0..3) and +16 (v4..7).
__device__ __forceinline__ v16bf load_frag(const unsigned short* lds, int rowOff, int lane) {
    int r  = rowOff + (lane & 15);
    int kb = (lane >> 4) * 8;
    const int4 c0 = *reinterpret_cast<const int4*>(lds + r * LDS_STRIDE + kb);
    const int4 c1 = *reinterpret_cast<const int4*>(lds + r * LDS_STRIDE + kb + 16);
    FragAB f;
    f.i[0] = c0.x; f.i[1] = c0.y; f.i[2] = c0.z; f.i[3] = c0.w;
    f.i[4] = c1.x; f.i[5] = c1.y; f.i[6] = c1.z; f.i[7] = c1.w;
    return f.b;
}

// Shared double-buffered WMMA mainloop: acc[2][4] covers 32(m) x 64(n) per wave.
__device__ __forceinline__ void gemm_mainloop(const unsigned short* __restrict__ A, int m0,
                                              const unsigned short* __restrict__ Bt, int n0,
                                              unsigned short (*As)[BM * LDS_STRIDE],
                                              unsigned short (*Bs)[BN * LDS_STRIDE],
                                              int tid, int lane, int mOff, int nOff,
                                              v8f acc[2][4]) {
    stage_async(A,  m0, 0, As[0], tid);
    stage_async(Bt, n0, 0, Bs[0], tid);
    wait_async_all();
    __syncthreads();

    int cur = 0;
    for (int kk = 0; kk < DD; kk += BK, cur ^= 1) {
        if (kk + BK < DD) {                     // overlap DMA of next slice
            stage_async(A,  m0, kk + BK, As[cur ^ 1], tid);
            stage_async(Bt, n0, kk + BK, Bs[cur ^ 1], tid);
        }
        const unsigned short* as = As[cur];
        const unsigned short* bs = Bs[cur];
        v16bf a0 = load_frag(as, mOff,      lane);
        v16bf a1 = load_frag(as, mOff + 16, lane);
        v16bf b0 = load_frag(bs, nOff,      lane);
        v16bf b1 = load_frag(bs, nOff + 16, lane);
        v16bf b2 = load_frag(bs, nOff + 32, lane);
        v16bf b3 = load_frag(bs, nOff + 48, lane);
        acc[0][0] = wmma_bf16(a0, b0, acc[0][0]);
        acc[0][1] = wmma_bf16(a0, b1, acc[0][1]);
        acc[0][2] = wmma_bf16(a0, b2, acc[0][2]);
        acc[0][3] = wmma_bf16(a0, b3, acc[0][3]);
        acc[1][0] = wmma_bf16(a1, b0, acc[1][0]);
        acc[1][1] = wmma_bf16(a1, b1, acc[1][1]);
        acc[1][2] = wmma_bf16(a1, b2, acc[1][2]);
        acc[1][3] = wmma_bf16(a1, b3, acc[1][3]);
        wait_async_all();                       // next buffer resident
        __syncthreads();                        // all waves done with cur buffer
    }
}

// ---------------- Kernel 1: F (f32) -> Fbf (bf16) ----------------
__global__ __launch_bounds__(256) void convert_F(const float* __restrict__ F,
                                                 unsigned short* __restrict__ Fbf) {
    size_t i = ((size_t)blockIdx.x * 256 + threadIdx.x) * 4;
    float4 v = *reinterpret_cast<const float4*>(F + i);
    Pack4 p;
    p.u[0] = f32_to_bf16(v.x); p.u[1] = f32_to_bf16(v.y);
    p.u[2] = f32_to_bf16(v.z); p.u[3] = f32_to_bf16(v.w);
    *reinterpret_cast<uint2*>(Fbf + i) = p.v;
}

// ---- Kernel 2: [X|Xq|0] (B,D,n) f32 -> XcT (B,NCP,D) bf16 transposed ----
__global__ __launch_bounds__(256) void convert_X(const float* __restrict__ X,
                                                 const float* __restrict__ Xq,
                                                 unsigned short* __restrict__ XcT) {
    int b  = blockIdx.z;
    int d0 = blockIdx.y * 32;
    int n0 = blockIdx.x * 32;
    int t  = threadIdx.x;
    int r  = t >> 3;
    int c4 = (t & 7) * 4;
    unsigned short* dst = XcT + (size_t)b * NCP * DD + (size_t)(n0 + r) * DD + d0 + c4;
    if (n0 >= NC) {                  // zero padding rows 576..639
        Pack4 p; p.u[0] = p.u[1] = p.u[2] = p.u[3] = 0;
        *reinterpret_cast<uint2*>(dst) = p.v;
        return;
    }
    __shared__ float tile[32][33];
    const float* src; int ld; int ns0;
    if (n0 < PP) { src = X  + (size_t)b * DD * PP; ld = PP; ns0 = n0; }
    else         { src = Xq + (size_t)b * DD * KK; ld = KK; ns0 = n0 - PP; }
    float4 v = *reinterpret_cast<const float4*>(src + (size_t)(d0 + r) * ld + ns0 + c4);
    tile[r][c4 + 0] = v.x; tile[r][c4 + 1] = v.y;
    tile[r][c4 + 2] = v.z; tile[r][c4 + 3] = v.w;
    __syncthreads();
    Pack4 p;
#pragma unroll
    for (int i = 0; i < 4; ++i) p.u[i] = f32_to_bf16(tile[c4 + i][r]);
    *reinterpret_cast<uint2*>(dst) = p.v;
}

// ------- Kernel 3: lift GEMM, dual transposed store: tXT and gamma*tXT -------
// C[m=d][n] = sum_k F[m][k]*XcT[n][k]; store tXT[n][d] and tXTg[n][d] (bf16).
__global__ __launch_bounds__(256) void gemm_lift(const unsigned short* __restrict__ Fbf,
                                                 const unsigned short* __restrict__ XcT,
                                                 const float* __restrict__ gamma,
                                                 unsigned short* __restrict__ tXT,
                                                 unsigned short* __restrict__ tXTg) {
    int b  = blockIdx.z;
    int m0 = blockIdx.y * BM;                              // d
    int n0 = blockIdx.x * BN;                              // n
    const unsigned short* A  = Fbf;                        // [D][D]
    const unsigned short* Bt = XcT + (size_t)b * NCP * DD; // [NCP][D]
    unsigned short* Ct  = tXT  + (size_t)b * NCP * DD;
    unsigned short* Ctg = tXTg + (size_t)b * NCP * DD;

    __shared__ unsigned short As[2][BM * LDS_STRIDE];
    __shared__ unsigned short Bs[2][BN * LDS_STRIDE];

    int tid = threadIdx.x, lane = tid & 31, w = tid >> 5;
    int mOff = (w & 3) * 32;
    int nOff = (w >> 2) * 64;

    v8f acc[2][4];
#pragma unroll
    for (int i = 0; i < 2; ++i)
#pragma unroll
        for (int j = 0; j < 4; ++j) acc[i][j] = v8f_zero();

    gemm_mainloop(A, m0, Bt, n0, As, Bs, tid, lane, mOff, nOff, acc);

    int nl = lane & 15, hi = lane >> 4;
#pragma unroll
    for (int i = 0; i < 2; ++i) {
        int mm = m0 + mOff + i * 16 + hi * 8;              // 8 contiguous d
        const float4 g0 = *reinterpret_cast<const float4*>(gamma + mm);
        const float4 g1 = *reinterpret_cast<const float4*>(gamma + mm + 4);
        float gv[8] = {g0.x, g0.y, g0.z, g0.w, g1.x, g1.y, g1.z, g1.w};
#pragma unroll
        for (int j = 0; j < 4; ++j) {
            int nn = n0 + nOff + j * 16 + nl;              // n row
            Pack8 p, pg;
#pragma unroll
            for (int r = 0; r < 8; ++r) {
                p.u[r]  = f32_to_bf16(acc[i][j][r]);
                pg.u[r] = f32_to_bf16(acc[i][j][r] * gv[r]);
            }
            *reinterpret_cast<int4*>(Ct  + (size_t)nn * DD + mm) = p.v;
            *reinterpret_cast<int4*>(Ctg + (size_t)nn * DD + mm) = pg.v;
        }
    }
}

// ------- Kernel 4: Gram GEMM  Call[b][n1][n2] = sum_d (γ·tXT)[n1][d]·tXT[n2][d] -----
// Rows 0..511 = G (PxP), rows 512..575 = H (KxP). f32 out.
__global__ __launch_bounds__(256) void gemm_gram(const unsigned short* __restrict__ tXTg,
                                                 const unsigned short* __restrict__ tXT,
                                                 float* __restrict__ Call) {
    int b  = blockIdx.z;
    int m0 = blockIdx.y * BM;                              // n1 in [0,640)
    int n0 = blockIdx.x * BN;                              // n2 in [0,512)
    const unsigned short* A  = tXTg + (size_t)b * NCP * DD;
    const unsigned short* Bt = tXT  + (size_t)b * NCP * DD;
    float* C = Call + (size_t)b * NCP * PP;

    __shared__ unsigned short As[2][BM * LDS_STRIDE];
    __shared__ unsigned short Bs[2][BN * LDS_STRIDE];

    int tid = threadIdx.x, lane = tid & 31, w = tid >> 5;
    int mOff = (w & 3) * 32;
    int nOff = (w >> 2) * 64;

    v8f acc[2][4];
#pragma unroll
    for (int i = 0; i < 2; ++i)
#pragma unroll
        for (int j = 0; j < 4; ++j) acc[i][j] = v8f_zero();

    gemm_mainloop(A, m0, Bt, n0, As, Bs, tid, lane, mOff, nOff, acc);

    int nl = lane & 15, hi = lane >> 4;
#pragma unroll
    for (int i = 0; i < 2; ++i) {
#pragma unroll
        for (int j = 0; j < 4; ++j) {
            int mBase = m0 + mOff + i * 16 + hi * 8;
            int nn    = n0 + nOff + j * 16 + nl;
#pragma unroll
            for (int r = 0; r < 8; ++r)
                C[(size_t)(mBase + r) * PP + nn] = acc[i][j][r];
        }
    }
}

// ------- Kernel 5: per-batch iterative solver, L2-resident G/H -------
__global__ __launch_bounds__(256) void solver(const float* __restrict__ y,
                                              const float* __restrict__ Call,
                                              float* __restrict__ out) {
    int b = blockIdx.x;
    int t = threadIdx.x;
    const float* G = Call + (size_t)b * NCP * PP;      // rows 0..511
    const float* H = G + (size_t)PP * PP;              // rows 512..575

    __shared__ float rs[PP];
    __shared__ float accs[PP];
    __shared__ float red[256];

    rs[t]         = y[(size_t)b * PP + t];
    rs[t + 256]   = y[(size_t)b * PP + t + 256];
    accs[t]       = 0.0f;
    accs[t + 256] = 0.0f;
    __syncthreads();

    const float invLP = 1.0f / ((float)LSTEPS * (float)PP);
    int p0 = t, p1 = t + 256;

    for (int step = 0; step < LSTEPS; ++step) {
        accs[p0] += rs[p0];
        accs[p1] += rs[p1];
        if (step == LSTEPS - 1) break;   // last r-update never consumed
        float g0 = 0.0f, g1 = 0.0f;
        const float4* G0 = reinterpret_cast<const float4*>(G + (size_t)p0 * PP);
        const float4* G1 = reinterpret_cast<const float4*>(G + (size_t)p1 * PP);
#pragma unroll 4
        for (int q4 = 0; q4 < PP / 4; ++q4) {
            float4 a = G0[q4];
            float4 c = G1[q4];
            float4 rv = *reinterpret_cast<const float4*>(rs + q4 * 4);
            g0 += a.x * rv.x + a.y * rv.y + a.z * rv.z + a.w * rv.w;
            g1 += c.x * rv.x + c.y * rv.y + c.z * rv.z + c.w * rv.w;
        }
        __syncthreads();                 // all dots done before r is rewritten
        rs[p0] -= g0 * invLP;
        rs[p1] -= g1 * invLP;
        __syncthreads();
    }
    __syncthreads();                     // accs complete before H·acc

    int row  = t >> 2;
    int part = t & 3;
    const float* Hr = H + (size_t)row * PP + part * 128;
    const float* av = accs + part * 128;
    float s = 0.0f;
#pragma unroll 4
    for (int q = 0; q < 128; q += 4)
        s += Hr[q] * av[q] + Hr[q + 1] * av[q + 1]
           + Hr[q + 2] * av[q + 2] + Hr[q + 3] * av[q + 3];
    red[t] = s;
    __syncthreads();
    if (part == 0)
        out[(size_t)b * KK + row] = (red[t] + red[t + 1] + red[t + 2] + red[t + 3]) * invLP;
}

extern "C" void kernel_launch(void* const* d_in, const int* in_sizes, int n_in,
                              void* d_out, int out_size, void* d_ws, size_t ws_size,
                              hipStream_t stream) {
    (void)in_sizes; (void)n_in; (void)out_size; (void)ws_size;
    const float* X     = (const float*)d_in[0];   // (B, D, P)
    const float* Xq    = (const float*)d_in[1];   // (B, D, K)
    const float* y     = (const float*)d_in[2];   // (B, P)
    const float* F     = (const float*)d_in[3];   // (D, D)
    const float* gamma = (const float*)d_in[4];   // (D,)
    float* out = (float*)d_out;                   // (B, K)

    char* ws = (char*)d_ws;
    const size_t SZ_F   = (size_t)DD * DD * 2;           //   2 MB  bf16 F
    const size_t SZ_BF  = (size_t)BSZ * NCP * DD * 2;    //  84 MB  per bf16 tensor
    unsigned short* Fbf  = (unsigned short*)(ws);
    unsigned short* XcT  = (unsigned short*)(ws + SZ_F);
    unsigned short* tXT  = (unsigned short*)(ws + SZ_F + SZ_BF);
    unsigned short* tXTg = (unsigned short*)(ws + SZ_F + 2 * SZ_BF);
    float*          Call = (float*)(ws + SZ_F + 3 * SZ_BF);   // 84 MB f32 [G;H;pad]

    convert_F<<<dim3((DD * DD) / (256 * 4)), 256, 0, stream>>>(F, Fbf);
    convert_X<<<dim3(NCP / 32, DD / 32, BSZ), 256, 0, stream>>>(X, Xq, XcT);
    gemm_lift<<<dim3(NCP / BN, DD / BM, BSZ), 256, 0, stream>>>(Fbf, XcT, gamma, tXT, tXTg);
    gemm_gram<<<dim3(PP / BN, NCP / BM, BSZ), 256, 0, stream>>>(tXTg, tXT, Call);
    solver<<<dim3(BSZ), 256, 0, stream>>>(y, Call, out);
}
